// SAGE_57372173140138
// MI455X (gfx1250) — compile-verified, matrix operand order checked by reading, not run on hardware
//
#include <hip/hip_runtime.h>

#define NN   100000
#define NE   1600000
#define DIN  128
#define DHID 128
#define DOUT 64

typedef __attribute__((ext_vector_type(16))) __bf16 v16bf;
typedef __attribute__((ext_vector_type(8)))  __bf16 v8bf;
typedef __attribute__((ext_vector_type(8)))  float  v8f;
typedef __attribute__((ext_vector_type(4)))  float  v4f;

// ---------------- utility kernels ----------------

__global__ void k_zero(float* __restrict__ p, long n) {
  long i = (long)blockIdx.x * blockDim.x + threadIdx.x;
  long s = (long)gridDim.x * blockDim.x;
  for (; i < n; i += s) p[i] = 0.0f;
}

__global__ void k_convw(const float* __restrict__ W1, const float* __restrict__ W2,
                        __bf16* __restrict__ W1b, __bf16* __restrict__ W2b) {
  int i = blockIdx.x * blockDim.x + threadIdx.x;
  if (i < DHID * DIN)  W1b[i] = (__bf16)W1[i];
  if (i < DOUT * DHID) W2b[i] = (__bf16)W2[i];
}

__global__ void k_deg(const int* __restrict__ col, float* __restrict__ deg) {
  int e = blockIdx.x * blockDim.x + threadIdx.x;
  if (e < NE) atomicAdd(&deg[col[e]], 1.0f);
}

// ---------------- edge scatter (segment-sum) ----------------
// layer 1: 128 floats / edge, 32 lanes x float4 per edge
__global__ void k_scatter1(const float* __restrict__ x, const int* __restrict__ row,
                           const int* __restrict__ col, float* __restrict__ msg) {
  long t = (long)blockIdx.x * blockDim.x + threadIdx.x;
  long e = t >> 5;
  int  c = (int)(t & 31) * 4;
  if (e >= NE) return;
  int r = row[e], d = col[e];
  v4f v = *(const v4f*)(x + (long)r * DIN + c);
  float* dst = msg + (long)d * DIN + c;
  atomicAdd(dst + 0, v[0]);
  atomicAdd(dst + 1, v[1]);
  atomicAdd(dst + 2, v[2]);
  atomicAdd(dst + 3, v[3]);
}

// layer 2: 64 floats / edge, 16 lanes x float4 per edge
__global__ void k_scatter2(const float* __restrict__ h2, const int* __restrict__ row,
                           const int* __restrict__ col, float* __restrict__ msg) {
  long t = (long)blockIdx.x * blockDim.x + threadIdx.x;
  long e = t >> 4;
  int  c = (int)(t & 15) * 4;
  if (e >= NE) return;
  int r = row[e], d = col[e];
  v4f v = *(const v4f*)(h2 + (long)r * DOUT + c);
  float* dst = msg + (long)d * DOUT + c;
  atomicAdd(dst + 0, v[0]);
  atomicAdd(dst + 1, v[1]);
  atomicAdd(dst + 2, v[2]);
  atomicAdd(dst + 3, v[3]);
}

// ---------------- GEMM1: h1 = relu( ((msg1+x)/deg) @ W1^T + b1 ), bf16 out ----
// one wave per 16-node tile; 8 accumulator tiles cover HID=128
__global__ void __launch_bounds__(256) k_gemm1(
    const float* __restrict__ x, const float* __restrict__ msg,
    const float* __restrict__ deg, const __bf16* __restrict__ W1b,
    const float* __restrict__ b1, __bf16* __restrict__ h1) {
  int wave = threadIdx.x >> 5;
  int lane = threadIdx.x & 31;
  int tile = blockIdx.x * 8 + wave;
  if (tile >= NN / 16) return;           // wave-uniform: EXEC stays all-ones
  int m0  = tile * 16;
  int l16 = lane & 15;
  int hi  = lane >> 4;
  int mrow = m0 + l16;
  float rdeg = 1.0f / (deg[mrow] + 1.0f);
  const float* xr = x   + (long)mrow * DIN;
  const float* mr = msg + (long)mrow * DIN;

  v8f acc[8];
#pragma unroll
  for (int t = 0; t < 8; ++t)
#pragma unroll
    for (int j = 0; j < 8; ++j) acc[t][j] = 0.0f;

#pragma unroll
  for (int kc = 0; kc < 4; ++kc) {
    int off0 = kc * 32 + hi * 8;         // A layout: elems 0-7 -> K=off0+j
    int off1 = off0 + 16;                //           elems 8-15 -> K=off1+j
    v16bf a;
#pragma unroll
    for (int j = 0; j < 8; ++j) {        // fused aggregation epilogue
      a[j]     = (__bf16)((mr[off0 + j] + xr[off0 + j]) * rdeg);
      a[j + 8] = (__bf16)((mr[off1 + j] + xr[off1 + j]) * rdeg);
    }
#pragma unroll
    for (int nt = 0; nt < 8; ++nt) {
      const __bf16* wr = W1b + (long)(nt * 16 + l16) * DIN;  // B[k][n] = W[n][k]
      v8bf blo = *(const v8bf*)(wr + off0);
      v8bf bhi = *(const v8bf*)(wr + off1);
      v16bf b;
#pragma unroll
      for (int j = 0; j < 8; ++j) { b[j] = blo[j]; b[j + 8] = bhi[j]; }
      acc[nt] = __builtin_amdgcn_wmma_f32_16x16x32_bf16(
          false, a, false, b, (short)0, acc[nt], false, false);
    }
  }

#pragma unroll
  for (int nt = 0; nt < 8; ++nt) {
    int n = nt * 16 + l16;
    float bias = b1[n];
#pragma unroll
    for (int r = 0; r < 8; ++r) {        // C/D: VGPR r -> rows r / r+8
      int m = m0 + hi * 8 + r;
      float v = acc[nt][r] + bias;
      h1[(long)m * DHID + n] = (__bf16)(v > 0.0f ? v : 0.0f);
    }
  }
}

// ---------------- GEMM2: h2 = h1 @ W2^T (f32 out, pre-aggregation) ----------
__global__ void __launch_bounds__(256) k_gemm2(
    const __bf16* __restrict__ h1, const __bf16* __restrict__ W2b,
    float* __restrict__ h2) {
  int wave = threadIdx.x >> 5;
  int lane = threadIdx.x & 31;
  int tile = blockIdx.x * 8 + wave;
  if (tile >= NN / 16) return;
  int m0  = tile * 16;
  int l16 = lane & 15;
  int hi  = lane >> 4;
  const __bf16* ar = h1 + (long)(m0 + l16) * DHID;

  v8f acc[4];
#pragma unroll
  for (int t = 0; t < 4; ++t)
#pragma unroll
    for (int j = 0; j < 8; ++j) acc[t][j] = 0.0f;

#pragma unroll
  for (int kc = 0; kc < 4; ++kc) {
    int off0 = kc * 32 + hi * 8;
    int off1 = off0 + 16;
    v8bf alo = *(const v8bf*)(ar + off0);
    v8bf ahi = *(const v8bf*)(ar + off1);
    v16bf a;
#pragma unroll
    for (int j = 0; j < 8; ++j) { a[j] = alo[j]; a[j + 8] = ahi[j]; }
#pragma unroll
    for (int nt = 0; nt < 4; ++nt) {
      const __bf16* wr = W2b + (long)(nt * 16 + l16) * DHID;
      v8bf blo = *(const v8bf*)(wr + off0);
      v8bf bhi = *(const v8bf*)(wr + off1);
      v16bf b;
#pragma unroll
      for (int j = 0; j < 8; ++j) { b[j] = blo[j]; b[j + 8] = bhi[j]; }
      acc[nt] = __builtin_amdgcn_wmma_f32_16x16x32_bf16(
          false, a, false, b, (short)0, acc[nt], false, false);
    }
  }

#pragma unroll
  for (int nt = 0; nt < 4; ++nt) {
    int n = nt * 16 + l16;
#pragma unroll
    for (int r = 0; r < 8; ++r) {
      int m = m0 + hi * 8 + r;
      h2[(long)m * DOUT + n] = acc[nt][r];
    }
  }
}

// ---------------- final: out = (msg2 + h2)/deg + b2 ----------------
__global__ void k_final(const float* __restrict__ msg2, const float* __restrict__ h2,
                        const float* __restrict__ deg, const float* __restrict__ b2,
                        float* __restrict__ out) {
  long i = (long)blockIdx.x * blockDim.x + threadIdx.x;
  if (i >= (long)NN * DOUT) return;
  int node = (int)(i >> 6);
  int d    = (int)(i & 63);
  out[i] = (msg2[i] + h2[i]) / (deg[node] + 1.0f) + b2[d];
}

// ---------------- launch ----------------
extern "C" void kernel_launch(void* const* d_in, const int* in_sizes, int n_in,
                              void* d_out, int out_size, void* d_ws, size_t ws_size,
                              hipStream_t stream) {
  const float* x  = (const float*)d_in[0];
  const float* W1 = (const float*)d_in[1];
  const float* b1 = (const float*)d_in[2];
  const float* W2 = (const float*)d_in[3];
  const float* b2 = (const float*)d_in[4];
  const int*   row = (const int*)d_in[5];
  const int*   col = (const int*)d_in[6];
  float* out = (float*)d_out;

  char* ws = (char*)d_ws;
  size_t o = 0;
  float*  deg  = (float*)(ws + o);  o += (size_t)NN * 4;            // contiguous:
  float*  msg1 = (float*)(ws + o);  o += (size_t)NN * DIN * 4;      //   deg|msg1|msg2
  float*  msg2 = (float*)(ws + o);  o += (size_t)NN * DOUT * 4;     //   zeroed as one run
  float*  h2   = (float*)(ws + o);  o += (size_t)NN * DOUT * 4;
  __bf16* h1   = (__bf16*)(ws + o); o += (size_t)NN * DHID * 2;
  __bf16* W1b  = (__bf16*)(ws + o); o += (size_t)DHID * DIN * 2;
  __bf16* W2b  = (__bf16*)(ws + o); o += (size_t)DOUT * DHID * 2;
  (void)ws_size; (void)in_sizes; (void)n_in; (void)out_size;

  long nz = (long)NN * (1 + DIN + DOUT);                 // deg+msg1+msg2
  k_zero<<<2048, 256, 0, stream>>>(deg, nz);
  k_convw<<<(DHID * DIN + 255) / 256, 256, 0, stream>>>(W1, W2, W1b, W2b);
  k_deg<<<(NE + 255) / 256, 256, 0, stream>>>(col, deg);
  k_scatter1<<<(int)(((long)NE * 32 + 255) / 256), 256, 0, stream>>>(x, row, col, msg1);
  int tiles = NN / 16;
  k_gemm1<<<(tiles + 7) / 8, 256, 0, stream>>>(x, msg1, deg, W1b, b1, h1);
  k_gemm2<<<(tiles + 7) / 8, 256, 0, stream>>>(h1, W2b, h2);
  k_scatter2<<<(int)(((long)NE * 16 + 255) / 256), 256, 0, stream>>>(h2, row, col, msg2);
  k_final<<<(int)(((long)NN * DOUT + 255) / 256), 256, 0, stream>>>(msg2, h2, deg, b2, out);
}
